// GTLayer_15994458211403
// MI455X (gfx1250) — matrix operand order repaired and checked by reading, not compiled, not run on hardware
//
#include <hip/hip_runtime.h>
#include <hip/hip_fp16.h>

#define NN   4096
#define NJ   5
#define NE   150000
#define NC   2

#define TILE_M 128
#define TILE_N 128
#define TILE_K 32
#define LDSP   40   // padded LDS row stride in halfs: 80 B (multiple of 16 B, bank-conflict-free)

typedef __attribute__((ext_vector_type(16))) _Float16 v16h;
typedef __attribute__((ext_vector_type(8)))  _Float16 v8h;
typedef __attribute__((ext_vector_type(8)))  float    v8f;

// ---------------- softmax over edge types: f = softmax(w, axis=1), w:[C,J] ----
__global__ void gt_softmax_kernel(const float* __restrict__ w1,
                                  const float* __restrict__ w2,
                                  float* __restrict__ fout /* f1[C][J] then f2[C][J] */) {
    int t = threadIdx.x;
    if (t >= 2 * NC) return;
    int sel = t >> 1;          // 0 -> w1/f1, 1 -> w2/f2
    int ch  = t & 1;
    const float* w = (sel ? w2 : w1) + ch * NJ;
    float m = w[0];
    for (int j = 1; j < NJ; ++j) m = fmaxf(m, w[j]);
    float e[NJ];
    float s = 0.f;
    for (int j = 0; j < NJ; ++j) { e[j] = expf(w[j] - m); s += e[j]; }
    float inv = 1.0f / s;
    for (int j = 0; j < NJ; ++j)
        fout[sel * NC * NJ + ch * NJ + j] = e[j] * inv;
}

// packed-f16 atomic accumulate of one half at linear index idx within a plane
__device__ __forceinline__ void pk_add_half(__half* base, size_t idx, float v) {
    __half2* p = (__half2*)(base + (idx & ~(size_t)1));     // 4B-aligned pair
    __half hv = __float2half(v);
    __half z  = __float2half(0.f);
    __half2 val = (idx & 1) ? __halves2half2(z, hv) : __halves2half2(hv, z);
    unsafeAtomicAdd(p, val);                                 // global_atomic_pk_add_f16
}

// ---------------- fused scatter into f16 planes; A2 stored TRANSPOSED --------
__global__ void gt_scatter_kernel(const int* __restrict__ ei,     // [J,2,E]
                                  const float* __restrict__ ev,   // [J,E]
                                  const float* __restrict__ f,    // f1[C][J], f2[C][J]
                                  __half* __restrict__ A1h,       // [C,N,N]  (row, col)
                                  __half* __restrict__ A2t) {     // [C,N,N]  (col, row)!
    int idx = blockIdx.x * blockDim.x + threadIdx.x;
    if (idx >= NJ * NE) return;
    int j = idx / NE;
    int e = idx - j * NE;
    int r = ei[(size_t)j * 2 * NE + e];
    int c = ei[(size_t)j * 2 * NE + NE + e];
    float v = ev[(size_t)j * NE + e];
#pragma unroll
    for (int ch = 0; ch < NC; ++ch) {
        pk_add_half(A1h + (size_t)ch * NN * NN, (size_t)r * NN + c, f[ch * NJ + j] * v);
        pk_add_half(A2t + (size_t)ch * NN * NN, (size_t)c * NN + r, f[NC * NJ + ch * NJ + j] * v);
    }
}

// ---------------- H[c] = A1[c] @ A2[c]; f16 operands, async-LDS double buffer -
__global__ __launch_bounds__(256) void gt_gemm_kernel(const _Float16* __restrict__ Aall,
                                                      const _Float16* __restrict__ Ball, // transposed [n][k]
                                                      float* __restrict__ Hall) {
    __shared__ __attribute__((aligned(16))) _Float16 As[2][TILE_M][LDSP]; // [buf][m][k]
    __shared__ __attribute__((aligned(16))) _Float16 Bs[2][TILE_N][LDSP]; // [buf][n][k]

    const int ch = blockIdx.z;
    const _Float16* A = Aall + (size_t)ch * NN * NN;
    const _Float16* B = Ball + (size_t)ch * NN * NN;
    float*          H = Hall + (size_t)ch * NN * NN;

    const int tile_m = blockIdx.y * TILE_M;
    const int tile_n = blockIdx.x * TILE_N;

    const int tid  = threadIdx.x;
    const int lane = tid & 31;
    const int wid  = tid >> 5;           // 0..7
    const int wm   = (wid >> 1) * 32;    // wave row offset: 0,32,64,96
    const int wn   = (wid & 1) * 64;     // wave col offset: 0,64
    const int lrow = lane & 15;
    const int g    = lane >> 4;          // lane half

    v8f acc[2][4];
#pragma unroll
    for (int mi = 0; mi < 2; ++mi)
#pragma unroll
        for (int ni = 0; ni < 4; ++ni)
            acc[mi][ni] = 0.0f;

    // async prefetch of one K-slab (A and B tiles, each 128 rows x 64 B) into buffer `buf`
    auto prefetch = [&](int buf, int k0) {
#pragma unroll
        for (int it = 0; it < 2; ++it) {
            int q   = tid + 256 * it;      // 0..511 : 16-byte chunks of the A tile
            int row = q >> 2;              // 0..127
            int c16 = (q & 3) << 3;        // half offset within row: 0,8,16,24
            const _Float16* ga = A + (size_t)(tile_m + row) * NN + k0 + c16;
            unsigned la = (unsigned)(uintptr_t)&As[buf][row][c16];
            asm volatile("global_load_async_to_lds_b128 %0, %1, off"
                         :: "v"(la), "v"(ga) : "memory");
            const _Float16* gb = B + (size_t)(tile_n + row) * NN + k0 + c16;
            unsigned lb = (unsigned)(uintptr_t)&Bs[buf][row][c16];
            asm volatile("global_load_async_to_lds_b128 %0, %1, off"
                         :: "v"(lb), "v"(gb) : "memory");
        }
    };

    prefetch(0, 0);
    asm volatile("s_wait_asynccnt 0x0" ::: "memory");
    __syncthreads();

    const int KB = NN / TILE_K;            // 128 K-slabs
    for (int kb = 0; kb < KB; ++kb) {
        const int cur = kb & 1;
        if (kb + 1 < KB)
            prefetch(cur ^ 1, (kb + 1) * TILE_K);   // overlaps with WMMA below

        // ---- fragments: contiguous K order (K = 16g + e), same for A and B ----
        // A 16x32: lane row M=lrow; elems e -> K = 16g + e
        v16h afrag[2];
#pragma unroll
        for (int mi = 0; mi < 2; ++mi) {
            int m = wm + 16 * mi + lrow;
            v8h lo = *(const v8h*)&As[cur][m][16 * g];
            v8h hi = *(const v8h*)&As[cur][m][16 * g + 8];
            v16h a;
#pragma unroll
            for (int q = 0; q < 8; ++q) { a[q] = lo[q]; a[q + 8] = hi[q]; }
            afrag[mi] = a;
        }
        // B 32x16 (column-major per lane): lane col N=lrow; elems e -> K = 16g + e
        v16h bfrag[4];
#pragma unroll
        for (int ni = 0; ni < 4; ++ni) {
            int n = wn + 16 * ni + lrow;
            v8h lo = *(const v8h*)&Bs[cur][n][16 * g];
            v8h hi = *(const v8h*)&Bs[cur][n][16 * g + 8];
            v16h b;
#pragma unroll
            for (int q = 0; q < 8; ++q) { b[q] = lo[q]; b[q + 8] = hi[q]; }
            bfrag[ni] = b;
        }

#pragma unroll
        for (int mi = 0; mi < 2; ++mi)
#pragma unroll
            for (int ni = 0; ni < 4; ++ni)
                acc[mi][ni] = __builtin_amdgcn_wmma_f32_16x16x32_f16(
                    false, afrag[mi], false, bfrag[ni],
                    (short)0, acc[mi][ni], false, false);

        // next buffer's async loads must be complete, and everyone done reading `cur`
        asm volatile("s_wait_asynccnt 0x0" ::: "memory");
        __syncthreads();
    }

    // ---- write back: C/D layout: VGPR r -> M = r + 8*g, N = lrow ----
#pragma unroll
    for (int mi = 0; mi < 2; ++mi) {
#pragma unroll
        for (int ni = 0; ni < 4; ++ni) {
            int rbase = tile_m + wm + 16 * mi + 8 * g;
            int col   = tile_n + wn + 16 * ni + lrow;
#pragma unroll
            for (int r = 0; r < 8; ++r)
                H[(size_t)(rbase + r) * NN + col] = acc[mi][ni][r];
        }
    }
}

extern "C" void kernel_launch(void* const* d_in, const int* in_sizes, int n_in,
                              void* d_out, int out_size, void* d_ws, size_t ws_size,
                              hipStream_t stream) {
    const int*   ei = (const int*)d_in[0];    // [J,2,E] int32
    const float* ev = (const float*)d_in[1];  // [J,E]
    const float* w1 = (const float*)d_in[2];  // [C,J]
    const float* w2 = (const float*)d_in[3];  // [C,J]

    float* H    = (float*)d_out;                       // [C,N,N]
    float* fout = H + (size_t)NC * NN * NN;            // f1[C,J], f2[C,J] in output tail

    __half* A1h = (__half*)d_ws;                       // [C,N,N] f16, row-major
    __half* A2t = A1h + (size_t)NC * NN * NN;          // [C,N,N] f16, TRANSPOSED [col][row]

    // zero the f16 accumulation buffers (128 MB) - capturable memset node
    hipMemsetAsync(d_ws, 0, (size_t)2 * NC * NN * NN * sizeof(__half), stream);

    gt_softmax_kernel<<<1, 32, 0, stream>>>(w1, w2, fout);

    int nthreads = NJ * NE;
    gt_scatter_kernel<<<(nthreads + 255) / 256, 256, 0, stream>>>(ei, ev, fout, A1h, A2t);

    dim3 grid(NN / TILE_N, NN / TILE_M, NC);
    gt_gemm_kernel<<<grid, 256, 0, stream>>>((const _Float16*)A1h, (const _Float16*)A2t, H);
}
